// CTR_6511170421700
// MI455X (gfx1250) — matrix operand order repaired and checked by reading, not compiled, not run on hardware
//
#include <hip/hip_runtime.h>
#include <hip/hip_bf16.h>

typedef __attribute__((ext_vector_type(16))) _Float16 v16h;
typedef __attribute__((ext_vector_type(8)))  _Float16 v8h;
typedef __attribute__((ext_vector_type(8)))  float    v8f;

#define WG_DIV_UP(a,b) (((a)+(b)-1)/(b))

__device__ __forceinline__ float leakyf(float x){ return x > 0.f ? x : 0.01f * x; }
__device__ __forceinline__ float sigmf(float x){ return 1.f / (1.f + __expf(-x)); }
__device__ __forceinline__ float softplusf(float x){ return x > 20.f ? x : log1pf(__expf(x)); }

// ---------------- WMMA fragment helpers (CDNA5 16x16x32 f16, wave32) -----------
// A fragment: 16x32 f16 tile (row-major LDS). Lane m (0..15) = row m, K in
// {0..15}; lane m+16 = row m, K in {16..31}. Two aligned 16B runs.
__device__ __forceinline__ v16h load_a_frag(const _Float16* A, int lda, int lane, int kb){
  const int row = lane & 15;
  const int hi  = (lane >> 4) & 1;
  const _Float16* p = A + row * lda + kb * 32 + hi * 16;
  v8h lo = *(const v8h*)p;
  v8h hh = *(const v8h*)(p + 8);
  return __builtin_shufflevector(lo, hh, 0,1,2,3,4,5,6,7,8,9,10,11,12,13,14,15);
}

// B fragments stored PRE-SWIZZLED in LDS: fragment (kb,nb) = 32 lanes x 16 halfs
// contiguous -> one aligned 32B vector load (2x ds_load_b128) per lane.
// element i of lane (kh=lane>>4, col=lane&15) is W[kb*32 + kh*16 + i][nb*16 + col].
template<int NB, bool TR>
__device__ __forceinline__ void stage_b(const float* __restrict__ Wsrc,
                                        _Float16* __restrict__ sB, int tid, int nthr){
  constexpr int NC = NB * 16;
  for (int idx = tid; idx < 64 * NC; idx += nthr){
    int i    = idx & 15;
    int lane = (idx >> 4) & 31;
    int frag = idx >> 9;           // kb*NB + nb
    int kb = frag / NB, nb = frag - kb * NB;
    int kh = lane >> 4, col = lane & 15;
    int k = kb * 32 + kh * 16 + i;
    int j = nb * 16 + col;
    float v = TR ? Wsrc[j * 64 + k] : Wsrc[k * NC + j];
    sB[idx] = (_Float16)v;
  }
}

// 16 rows x (NB*16) cols = A[16x64] @ B[64 x NB*16], accumulated into acc[NB].
template<int NB>
__device__ __forceinline__ void wave_gemm(const _Float16* A, int lda,
                                          const _Float16* sB, int lane, v8f* acc){
#pragma unroll
  for (int kb = 0; kb < 2; ++kb){
    v16h a = load_a_frag(A, lda, lane, kb);
#pragma unroll
    for (int nb = 0; nb < NB; ++nb){
      v16h b = *(const v16h*)(sB + ((((kb * NB + nb) << 5) + lane) << 4));
      acc[nb] = __builtin_amdgcn_wmma_f32_16x16x32_f16(false, a, false, b,
                                                       (short)0, acc[nb], false, false);
    }
  }
}

// ---------------- utility kernels ----------------------------------------------
__global__ void k_zero(float* p, long long n){
  long long i = (long long)blockIdx.x * blockDim.x + threadIdx.x;
  if (i < n) p[i] = 0.f;
}

__global__ void k_deg(const int* __restrict__ col, float* __restrict__ deg, long long E){
  long long e = (long long)blockIdx.x * blockDim.x + threadIdx.x;
  if (e < E) atomicAdd(&deg[col[e]], 1.0f);
}

// val = exp(-exp(-d^2)^2) * dinv[row]*dinv[col]
__global__ void k_val(const int* __restrict__ row, const int* __restrict__ col,
                      const float* __restrict__ dist, const float* __restrict__ deg,
                      float* __restrict__ val, long long E){
  long long e = (long long)blockIdx.x * blockDim.x + threadIdx.x;
  if (e >= E) return;
  float d  = dist[e];
  float t  = __expf(-d * d);
  float dw = __expf(-t * t);
  float dr = deg[row[e]], dc = deg[col[e]];
  float ir = dr > 0.f ? rsqrtf(dr) : 0.f;
  float ic = dc > 0.f ? rsqrtf(dc) : 0.f;
  val[e] = dw * ir * ic;
}

// out[dst[e]] += (val ? val[e] : 1) * mat[src[e]]   (D=64, 16 threads/edge, float4)
__global__ void k_scatter(const int* __restrict__ dst, const int* __restrict__ src,
                          const float* __restrict__ val, const float* __restrict__ mat,
                          float* __restrict__ out, long long E){
  long long idx = (long long)blockIdx.x * blockDim.x + threadIdx.x;
  long long e = idx >> 4;
  if (e >= E) return;
  int c = (int)(idx & 15) * 4;
  float v = val ? val[e] : 1.0f;
  const float4 s = *(const float4*)(mat + (long long)src[e] * 64 + c);
  float* o = out + (long long)dst[e] * 64 + c;
  atomicAdd(o + 0, v * s.x);
  atomicAdd(o + 1, v * s.y);
  atomicAdd(o + 2, v * s.z);
  atomicAdd(o + 3, v * s.w);
}

__global__ void k_neigh_div(float* __restrict__ neigh, const float* __restrict__ deg, long long n64){
  long long i = (long long)blockIdx.x * blockDim.x + threadIdx.x;
  if (i < n64) neigh[i] /= fmaxf(deg[i >> 6], 1.0f);
}

__global__ void k_counts(const int* __restrict__ bidx, float* __restrict__ counts, int T){
  int t = blockIdx.x * blockDim.x + threadIdx.x;
  if (t < T) atomicAdd(&counts[bidx[t]], 1.0f);
}

// pooled_neigh[b] = sum_t neigh[sess_idx[t]] / counts;  pooled_emb likewise
__global__ void k_pool(const int* __restrict__ sess_idx, const float* __restrict__ neigh,
                       const float* __restrict__ emb, const float* __restrict__ counts,
                       float* __restrict__ pn, float* __restrict__ pe, int L){
  int b = blockIdx.x, d = threadIdx.x;
  float s1 = 0.f, s2 = 0.f;
  int base = b * L;
  for (int t = 0; t < L; ++t){
    long long r = (long long)sess_idx[base + t] * 64;
    s1 += neigh[r + d];
    s2 += emb[r + d];
  }
  float c = counts[b];
  pn[(long long)b * 64 + d] = s1 / c;
  pe[(long long)b * 64 + d] = s2 / c;
}

// ---------------- WMMA GEMM kernels --------------------------------------------
// out[R x 64] = (gather? A[gather[r]] : A[r]) @ W[64x64].  REQUIRES R % 64 == 0.
__global__ void k_gemm_rows(const float* __restrict__ A, const int* __restrict__ gather,
                            const float* __restrict__ W, float* __restrict__ out, int R){
  __shared__ __attribute__((aligned(32))) _Float16 sW[64 * 64];
  __shared__ __attribute__((aligned(32))) _Float16 sA[4][16 * 64];
  __shared__ __attribute__((aligned(16))) float    sO[4][16 * 64];
  const int tid = threadIdx.x, lane = tid & 31, w = tid >> 5;
  stage_b<4, false>(W, sW, tid, 128);
  const int row0 = blockIdx.x * 64 + w * 16;
  for (int i = lane; i < 16 * 64; i += 32){
    int r = i >> 6, c = i & 63;
    long long s = gather ? (long long)gather[row0 + r] : (long long)(row0 + r);
    sA[w][i] = (_Float16)A[s * 64 + c];
  }
  __syncthreads();
  v8f acc[4] = {};
  wave_gemm<4>(&sA[w][0], 64, sW, lane, acc);
  const int cl = lane & 15, hi8 = ((lane >> 4) & 1) * 8;
#pragma unroll
  for (int nb = 0; nb < 4; ++nb)
#pragma unroll
    for (int v = 0; v < 8; ++v)
      sO[w][(hi8 + v) * 64 + nb * 16 + cl] = acc[nb][v];
  __syncthreads();
  // sO linear layout == out block layout -> coalesced b128 stores
  const float4* src = (const float4*)&sO[0][0];
  float4* dst = (float4*)(out + (long long)blockIdx.x * 64 * 64);
  for (int i = tid; i < 1024; i += 128) dst[i] = src[i];
}

// GCN layer: out = L2norm_rows( leaky( side@W0 + (feat*side)@W1 ) )
__global__ void k_gcn_layer(const float* __restrict__ feat, const float* __restrict__ side,
                            const float* __restrict__ W0, const float* __restrict__ W1,
                            float* __restrict__ out, int N){
  __shared__ __attribute__((aligned(32))) _Float16 sW0[4096], sW1[4096];
  __shared__ __attribute__((aligned(32))) _Float16 sS[4][1024], sFS[4][1024];
  __shared__ __attribute__((aligned(16))) float    sO[4][1024];
  __shared__ float rss[64][2];
  const int tid = threadIdx.x, lane = tid & 31, w = tid >> 5;
  stage_b<4, false>(W0, sW0, tid, 128);
  stage_b<4, false>(W1, sW1, tid, 128);
  const int row0 = blockIdx.x * 64 + w * 16;
  for (int i = lane; i < 1024; i += 32){
    int r = i >> 6, c = i & 63, gr = row0 + r;
    float s = 0.f, f = 0.f;
    if (gr < N){ long long o = (long long)gr * 64 + c; s = side[o]; f = feat[o]; }
    sS[w][i]  = (_Float16)s;
    sFS[w][i] = (_Float16)(f * s);
  }
  __syncthreads();
  v8f acc[4] = {};
  wave_gemm<4>(&sS[w][0],  64, sW0, lane, acc);
  wave_gemm<4>(&sFS[w][0], 64, sW1, lane, acc);
  const int cl = lane & 15, hi8 = ((lane >> 4) & 1) * 8;
#pragma unroll
  for (int nb = 0; nb < 4; ++nb)
#pragma unroll
    for (int v = 0; v < 8; ++v)
      sO[w][(hi8 + v) * 64 + nb * 16 + cl] = leakyf(acc[nb][v]);
  __syncthreads();
  // normalize: 2 threads per row, float4 everywhere
  const int row = tid >> 1, half = tid & 1;
  const float4* rp = (const float4*)&sO[row >> 4][(row & 15) * 64 + half * 32];
  float4 vv[8];
  float ss = 0.f;
#pragma unroll
  for (int i = 0; i < 8; ++i){
    vv[i] = rp[i];
    ss += vv[i].x * vv[i].x + vv[i].y * vv[i].y + vv[i].z * vv[i].z + vv[i].w * vv[i].w;
  }
  rss[row][half] = ss;
  __syncthreads();
  float inv = 1.0f / fmaxf(sqrtf(rss[row][0] + rss[row][1]), 1e-12f);
  int gr = blockIdx.x * 64 + row;
  if (gr < N){
    float4* op = (float4*)(out + (long long)gr * 64 + half * 32);
#pragma unroll
    for (int i = 0; i < 8; ++i){
      float4 t = vv[i];
      t.x *= inv; t.y *= inv; t.z *= inv; t.w *= inv;
      op[i] = t;
    }
  }
}

// Fused GatedGraphConv GRU cell:
//   gi = m@wih^T + bih ; gh = h@whh^T + bhh ; gates ; hid = leaky((1-z)n + z h)
// 2 waves / block, 32 rows / block.  REQUIRES T % 32 == 0.
__global__ void k_gru(const int* __restrict__ sess_idx, const float* __restrict__ poi,
                      const float* __restrict__ mbuf,
                      const float* __restrict__ wih, const float* __restrict__ whh,
                      const float* __restrict__ bih, const float* __restrict__ bhh,
                      float* __restrict__ hid, int T){
  __shared__ __attribute__((aligned(32))) _Float16 sBi[64 * 192], sBh[64 * 192];
  __shared__ __attribute__((aligned(32))) _Float16 sAm[2][1024], sAh[2][1024];
  const int tid = threadIdx.x, lane = tid & 31, w = tid >> 5;
  stage_b<12, true>(wih, sBi, tid, 64);
  stage_b<12, true>(whh, sBh, tid, 64);
  const int row0 = blockIdx.x * 32 + w * 16;
  for (int i = lane; i < 1024; i += 32){
    int r = i >> 6, c = i & 63, gr = row0 + r;
    sAm[w][i] = (_Float16)mbuf[(long long)gr * 64 + c];
    sAh[w][i] = (_Float16)poi[(long long)sess_idx[gr] * 64 + c];
  }
  __syncthreads();
  v8f ai[12] = {}, ah[12] = {};
  wave_gemm<12>(&sAm[w][0], 64, sBi, lane, ai);
  wave_gemm<12>(&sAh[w][0], 64, sBh, lane, ah);
  const int cl = lane & 15, hi8 = ((lane >> 4) & 1) * 8;
#pragma unroll
  for (int q = 0; q < 4; ++q)
#pragma unroll
    for (int v = 0; v < 8; ++v){
      int jc = q * 16 + cl;
      int lr = hi8 + v;
      float rg = sigmf(ai[q][v]     + bih[jc]       + ah[q][v]     + bhh[jc]);
      float zg = sigmf(ai[4 + q][v] + bih[64 + jc]  + ah[4 + q][v] + bhh[64 + jc]);
      float ng = tanhf(ai[8 + q][v] + bih[128 + jc] + rg * (ah[8 + q][v] + bhh[128 + jc]));
      float h  = (float)sAh[w][lr * 64 + jc];
      hid[(long long)(row0 + lr) * 64 + jc] = leakyf((1.f - zg) * ng + zg * h);
    }
}

// _seq_query: one block per session (64 contiguous rows). enc[b] = sum_t w_t * sess_t,
// w_t = sigmoid(qW1+b1 + sess@W2 + b2) @ a + ab
__global__ void k_seqquery(const float* __restrict__ sess_src, const int* __restrict__ sess_gather,
                           const float* __restrict__ query_src, const int* __restrict__ tar_poi,
                           const float* __restrict__ W1, const float* __restrict__ b1,
                           const float* __restrict__ W2, const float* __restrict__ b2,
                           const float* __restrict__ avec, const float* __restrict__ ab,
                           float* __restrict__ enc){
  const int b = blockIdx.x;
  __shared__ __attribute__((aligned(32))) _Float16 sW2[4096];
  __shared__ __attribute__((aligned(32))) _Float16 sS[4096];
  __shared__ __attribute__((aligned(16))) float sSf[4096];
  __shared__ float qv[64], sa[64], red[64 * 16], wvec[64];
  const int tid = threadIdx.x, lane = tid & 31, w = tid >> 5;
  stage_b<4, false>(W2, sW2, tid, 128);
  for (int i = tid; i < 4096; i += 128){
    int r = i >> 6, c = i & 63;
    int t = b * 64 + r;
    long long s = sess_gather ? (long long)sess_gather[t] : (long long)t;
    float v = sess_src[s * 64 + c];
    sSf[i] = v;
    sS[i]  = (_Float16)v;
  }
  if (tid < 64){
    int d = tid;
    long long qb = (long long)tar_poi[b] * 64;
    float s = b1[d] + b2[d];
    for (int k = 0; k < 64; ++k) s += query_src[qb + k] * W1[k * 64 + d];
    qv[d] = s;
    sa[d] = avec[d];
  }
  __syncthreads();
  v8f acc[4] = {};
  wave_gemm<4>(&sS[w * 1024], 64, sW2, lane, acc);
  const int cl = lane & 15, hi8 = ((lane >> 4) & 1) * 8;
  float part[8];
#pragma unroll
  for (int v = 0; v < 8; ++v) part[v] = 0.f;
#pragma unroll
  for (int nb = 0; nb < 4; ++nb)
#pragma unroll
    for (int v = 0; v < 8; ++v){
      int col = nb * 16 + cl;
      part[v] += sigmf(acc[nb][v] + qv[col]) * sa[col];
    }
#pragma unroll
  for (int v = 0; v < 8; ++v) red[(w * 16 + hi8 + v) * 16 + cl] = part[v];
  __syncthreads();
  if (tid < 64){
    float s = ab[0];
    for (int l = 0; l < 16; ++l) s += red[tid * 16 + l];
    wvec[tid] = s;
  }
  __syncthreads();
  if (tid < 64){
    int d = tid;
    float s = 0.f;
    for (int t = 0; t < 64; ++t) s += wvec[t] * sSf[t * 64 + d];
    enc[(long long)b * 64 + d] = s;
  }
}

// final: tar_embed, tar_geo_embed, y, con_loss (BPR), MLP logits
__global__ void k_final(const int* __restrict__ tar_poi, const float* __restrict__ y,
                        const float* __restrict__ poi, const float* __restrict__ geo_feat,
                        const float* __restrict__ geo_enc, const float* __restrict__ sess_enc,
                        const float* __restrict__ geo_pool, const float* __restrict__ sess_pool,
                        const float* __restrict__ geo_enc_p, const float* __restrict__ sess_enc_p,
                        const float* __restrict__ mlpW1, const float* __restrict__ mlpb1,
                        const float* __restrict__ mlpW2, const float* __restrict__ mlpb2,
                        float* __restrict__ o_logits, float* __restrict__ o_con,
                        float* __restrict__ o_y, float* __restrict__ o_tar,
                        float* __restrict__ o_targeo){
  const int b = blockIdx.x, d = threadIdx.x;
  __shared__ float feats[256];
  __shared__ float r4[4][64];
  __shared__ float redb[64];
  long long tp = (long long)tar_poi[b] * 64;
  long long bo = (long long)b * 64;
  float te = poi[tp + d], tg = geo_feat[tp + d];
  o_tar[bo + d] = te;
  o_targeo[bo + d] = tg;
  feats[d]       = te;
  feats[64 + d]  = tg;
  feats[128 + d] = sess_enc_p[bo + d];
  feats[192 + d] = geo_enc_p[bo + d];
  float ge = geo_enc[bo + d], se = sess_enc[bo + d];
  float gp = geo_pool[bo + d], sp = sess_pool[bo + d];
  r4[0][d] = ge * sp;
  r4[1][d] = ge * gp;
  r4[2][d] = se * gp;
  r4[3][d] = se * sp;
  __syncthreads();
  float s = mlpb1[d];
  for (int k = 0; k < 256; ++k) s += feats[k] * mlpW1[k * 64 + d];
  redb[d] = leakyf(s) * mlpW2[d];
  __syncthreads();
  if (d == 0){
    float s0 = 0.f, s1 = 0.f, s2 = 0.f, s3 = 0.f, sm = 0.f;
    for (int i = 0; i < 64; ++i){
      s0 += r4[0][i]; s1 += r4[1][i]; s2 += r4[2][i]; s3 += r4[3][i]; sm += redb[i];
    }
    o_con[b]    = softplusf(s0 - s1) + softplusf(s2 - s3);
    o_logits[b] = sm + mlpb2[0];
    o_y[b]      = y[b];
  }
}

// ---------------- launch --------------------------------------------------------
extern "C" void kernel_launch(void* const* d_in, const int* in_sizes, int n_in,
                              void* d_out, int out_size, void* d_ws, size_t ws_size,
                              hipStream_t stream) {
  const int*   sess_idx  = (const int*)  d_in[0];
  const int*   edges     = (const int*)  d_in[1];   // [2,T]
  const int*   batch_idx = (const int*)  d_in[2];
  const int*   tar_poi   = (const int*)  d_in[3];
  const float* yv        = (const float*)d_in[4];
  const int*   edge_idx  = (const int*)  d_in[5];   // [2,E]
  const float* dist_vec  = (const float*)d_in[6];
  const float* poi       = (const float*)d_in[7];
  const float* geoW0     = (const float*)d_in[8];
  const float* geoW1     = (const float*)d_in[9];
  const float* ggcW      = (const float*)d_in[10];
  const float* wih       = (const float*)d_in[11];
  const float* whh       = (const float*)d_in[12];
  const float* bih       = (const float*)d_in[13];
  const float* bhh       = (const float*)d_in[14];
  const float* gqW1 = (const float*)d_in[15]; const float* gqb1 = (const float*)d_in[16];
  const float* gqW2 = (const float*)d_in[17]; const float* gqb2 = (const float*)d_in[18];
  const float* gqa  = (const float*)d_in[19]; const float* gqab = (const float*)d_in[20];
  const float* sqW1 = (const float*)d_in[21]; const float* sqb1 = (const float*)d_in[22];
  const float* sqW2 = (const float*)d_in[23]; const float* sqb2 = (const float*)d_in[24];
  const float* sqa  = (const float*)d_in[25]; const float* sqab = (const float*)d_in[26];
  const float* geo_proj  = (const float*)d_in[27];
  const float* sess_proj = (const float*)d_in[28];
  const float* mlpW1 = (const float*)d_in[29]; const float* mlpb1 = (const float*)d_in[30];
  const float* mlpW2 = (const float*)d_in[31]; const float* mlpb2 = (const float*)d_in[32];

  const int       T = in_sizes[0];
  const int       B = in_sizes[3];
  const long long E = in_sizes[6];
  const int       N = in_sizes[7] / 64;
  const int       L = T / B;
  const long long N64 = (long long)N * 64;
  const long long T64 = (long long)T * 64;

  const int* e_row = edge_idx;
  const int* e_col = edge_idx + E;

  // workspace layout (floats)
  float* W = (float*)d_ws;
  size_t off = 0;
  auto alloc = [&](long long n)->float*{ float* p = W + off; off += (size_t)((n + 63) & ~63LL); return p; };
  float* deg      = alloc(N);
  float* val      = alloc(E);
  float* bufA     = alloc(N64);   // GCN layer0 out; later reused as hW
  float* bufB     = alloc(N64);   // GCN layer1 out (final geo_feat)
  float* side     = alloc(N64);   // scatter accumulator; later reused as neigh
  float* mbuf     = alloc(T64);
  float* hid      = alloc(T64);
  float* geo_enc  = alloc((long long)B * 64);
  float* sess_enc = alloc((long long)B * 64);
  float* pool_n   = alloc((long long)B * 64);
  float* pool_e   = alloc((long long)B * 64);
  float* geo_pl   = alloc((long long)B * 64);
  float* sess_pl  = alloc((long long)B * 64);
  float* counts   = alloc(B);
  (void)ws_size; (void)n_in; (void)out_size;

  float* out      = (float*)d_out;
  float* o_logits = out;
  float* o_con    = out + B;
  float* o_y      = out + 2 * B;
  float* o_tar    = out + 3 * B;
  float* o_targeo = o_tar + (long long)B * 64;
  float* o_geop   = o_targeo + (long long)B * 64;
  float* o_sessp  = o_geop + (long long)B * 64;

  auto zero = [&](float* p, long long n){
    k_zero<<<(unsigned)WG_DIV_UP(n, 256), 256, 0, stream>>>(p, n);
  };

  // degrees & edge weights
  zero(deg, N);
  k_deg<<<(unsigned)WG_DIV_UP(E, 256), 256, 0, stream>>>(e_col, deg, E);
  k_val<<<(unsigned)WG_DIV_UP(E, 256), 256, 0, stream>>>(e_row, e_col, dist_vec, deg, val, E);

  const unsigned scatE = (unsigned)WG_DIV_UP(E * 16, 256);
  const unsigned gRowsN = (unsigned)WG_DIV_UP(N, 64);
  const unsigned gRowsT = (unsigned)WG_DIV_UP(T, 64);
  const unsigned gRowsB = (unsigned)WG_DIV_UP(B, 64);

  // Geo GCN layer 0: side = seg_sum(val * poi[col], row)
  zero(side, N64);
  k_scatter<<<scatE, 256, 0, stream>>>(e_row, e_col, val, poi, side, E);
  k_gcn_layer<<<gRowsN, 128, 0, stream>>>(poi, side, geoW0, geoW1, bufA, N);
  // layer 1
  zero(side, N64);
  k_scatter<<<scatE, 256, 0, stream>>>(e_row, e_col, val, bufA, side, E);
  k_gcn_layer<<<gRowsN, 128, 0, stream>>>(bufA, side, geoW0 + 4096, geoW1 + 4096, bufB, N);

  // neighbor pooling (reuse side buffer as neigh)
  float* neigh = side;
  zero(neigh, N64);
  k_scatter<<<scatE, 256, 0, stream>>>(e_col, e_row, (const float*)nullptr, poi, neigh, E);
  k_neigh_div<<<(unsigned)WG_DIV_UP(N64, 256), 256, 0, stream>>>(neigh, deg, N64);

  zero(counts, B);
  k_counts<<<(unsigned)WG_DIV_UP(T, 256), 256, 0, stream>>>(batch_idx, counts, T);
  k_pool<<<B, 64, 0, stream>>>(sess_idx, neigh, poi, counts, pool_n, pool_e, L);

  // GatedGraphConv: hW = poi[sess_idx] @ ggc_W (reuse bufA), m = scatter, fused GRU
  float* hW = bufA;
  k_gemm_rows<<<gRowsT, 128, 0, stream>>>(poi, sess_idx, ggcW, hW, T);
  zero(mbuf, T64);
  k_scatter<<<(unsigned)WG_DIV_UP((long long)T * 16, 256), 256, 0, stream>>>(
      edges + T, edges, (const float*)nullptr, hW, mbuf, (long long)T);
  k_gru<<<(unsigned)WG_DIV_UP(T, 32), 64, 0, stream>>>(sess_idx, poi, mbuf,
                                                       wih, whh, bih, bhh, hid, T);

  // attention encoders
  k_seqquery<<<B, 128, 0, stream>>>(bufB, sess_idx, bufB, tar_poi,
                                    gqW1, gqb1, gqW2, gqb2, gqa, gqab, geo_enc);
  k_seqquery<<<B, 128, 0, stream>>>(hid, (const int*)nullptr, poi, tar_poi,
                                    sqW1, sqb1, sqW2, sqb2, sqa, sqab, sess_enc);

  // projections (WMMA)
  k_gemm_rows<<<gRowsB, 128, 0, stream>>>(geo_enc,  (const int*)nullptr, geo_proj,  o_geop,  B);
  k_gemm_rows<<<gRowsB, 128, 0, stream>>>(sess_enc, (const int*)nullptr, sess_proj, o_sessp, B);
  k_gemm_rows<<<gRowsB, 128, 0, stream>>>(pool_n,   (const int*)nullptr, geo_proj,  geo_pl,  B);
  k_gemm_rows<<<gRowsB, 128, 0, stream>>>(pool_e,   (const int*)nullptr, sess_proj, sess_pl, B);

  // BPR + MLP + output assembly
  k_final<<<B, 64, 0, stream>>>(tar_poi, yv, poi, bufB, geo_enc, sess_enc, geo_pl, sess_pl,
                                o_geop, o_sessp, mlpW1, mlpb1, mlpW2, mlpb2,
                                o_logits, o_con, o_y, o_tar, o_targeo);
}
